// GraphConvLayer_46110768890423
// MI455X (gfx1250) — matrix operand order repaired and checked by reading, not compiled, not run on hardware
//
#include <hip/hip_runtime.h>
#include <hip/hip_bf16.h>
#include <stdint.h>
#include <stddef.h>

// Problem constants (match reference)
#define B_   4
#define CIN  32
#define T_   8
#define N_   2000
#define COUT 64
#define E_   16000
#define H1_  128
#define H2_  256
#define BT   (B_ * T_)        // 32
#define M_   (BT * N_)        // 64000 rows, multiple of 128

typedef __attribute__((ext_vector_type(16))) __bf16 v16bf;
typedef __attribute__((ext_vector_type(2)))  __bf16 bf16x2;
typedef __attribute__((ext_vector_type(8)))  float  v8f;

// ---------------- degree / norm precompute ----------------

__global__ void k_deg_init(float* __restrict__ deg) {
    int n = blockIdx.x * blockDim.x + threadIdx.x;
    if (n < N_) deg[n] = 1.0f;   // self-loop weight
}

__global__ void k_deg_scatter(const long long* __restrict__ ei,
                              const float* __restrict__ ew,
                              float* __restrict__ deg) {
    int e = blockIdx.x * blockDim.x + threadIdx.x;
    if (e < E_) atomicAdd(&deg[(int)ei[E_ + e]], ew[e]);
}

__global__ void k_dinv(float* __restrict__ deg) {
    int n = blockIdx.x * blockDim.x + threadIdx.x;
    if (n < N_) deg[n] = rsqrtf(deg[n]);
}

__global__ void k_norm(const long long* __restrict__ ei,
                       const float* __restrict__ ew,
                       const float* __restrict__ dinv,
                       float* __restrict__ nrm) {
    int e = blockIdx.x * blockDim.x + threadIdx.x;
    if (e < E_) nrm[e] = dinv[(int)ei[e]] * ew[e] * dinv[(int)ei[E_ + e]];
}

// ---------------- layout transforms ----------------

// x[B,CIN,T,N] -> xt[(b*T+t)*N+n, CIN]
__global__ void k_transpose_x(const float* __restrict__ x, float* __restrict__ xt) {
    int id = blockIdx.x * blockDim.x + threadIdx.x;
    if (id >= B_ * CIN * T_ * N_) return;
    int n = id % N_;
    int t = (id / N_) % T_;
    int c = (id / (N_ * T_)) % CIN;
    int b = id / (N_ * T_ * CIN);
    xt[(((size_t)(b * T_ + t)) * N_ + n) * CIN + c] = x[id];
}

// W[K,Nc] f32 -> Wt[Nc,K] bf16 (so B-operand K-pairs are contiguous)
__global__ void k_wt_bf16(const float* __restrict__ W, __bf16* __restrict__ Wt,
                          int K, int Nc) {
    int id = blockIdx.x * blockDim.x + threadIdx.x;
    if (id >= K * Nc) return;
    int k = id / Nc;
    int n = id % Nc;
    Wt[(size_t)n * K + k] = (__bf16)W[id];
}

// ---------------- graph propagation ----------------

// D = dinv[n]^2 * S   (self-loop term; fully initializes D)
__global__ void k_prop_init(const float* __restrict__ S, float* __restrict__ D,
                            const float* __restrict__ dinv, int C) {
    size_t id = (size_t)blockIdx.x * blockDim.x + threadIdx.x;
    size_t total = (size_t)M_ * C;
    if (id >= total) return;
    int n = (int)((id / C) % N_);
    float dv = dinv[n];
    D[id] = dv * dv * S[id];
}

// one wave per (edge, bt) pair; lane strides channels; L2 fp32 atomics
__global__ void k_prop_scatter(const float* __restrict__ S, float* __restrict__ D,
                               const float* __restrict__ nrm,
                               const long long* __restrict__ ei, int C) {
    int wave = blockIdx.x * (blockDim.x >> 5) + (threadIdx.x >> 5);
    int lane = threadIdx.x & 31;
    if (wave >= E_ * BT) return;
    int e  = wave % E_;
    int bt = wave / E_;
    int s = (int)ei[e];
    int d = (int)ei[E_ + e];
    float w = nrm[e];
    const float* srow = S + ((size_t)bt * N_ + s) * C;
    float*       drow = D + ((size_t)bt * N_ + d) * C;
    for (int c = lane; c < C; c += 32)
        atomicAdd(&drow[c], w * srow[c]);
}

// ---------------- WMMA GEMM ----------------
// D[M,Nc] = A[M,K](f32, converted inline to bf16) x Wt[Nc,K](bf16) + bias
// 8 waves / block; each wave computes a 16x(16*TN) strip: A loaded & converted
// once per K-chunk, then TN back-to-back v_wmma_f32_16x16x32_bf16 issues.

template<int K, int TN>
__global__ void __launch_bounds__(256)
k_gemm_wmma(const float* __restrict__ A, const __bf16* __restrict__ Bt,
            const float* __restrict__ bias, float* __restrict__ D, int Nc) {
    const int lane = threadIdx.x & 31;
    const int wid  = threadIdx.x >> 5;
    const int tileM   = blockIdx.x * 8 + wid;
    const int colBase = blockIdx.y * (16 * TN) + (lane & 15);
    const int row = tileM * 16 + (lane & 15);
    const int hi  = lane >> 4;          // lane half
    const int kaA = hi * 8;             // A-matrix K offset for upper lanes
    const int kaB = hi * 8;             // B-matrix K offset for upper lanes

    v8f acc[TN];
#pragma unroll
    for (int t = 0; t < TN; ++t) acc[t] = (v8f){};

#pragma unroll
    for (int kb = 0; kb < K; kb += 32) {
        // A 16x32 bf16 layout: VGPR v holds K = (v>=4?16:0) + half*8 + 2*(v&3)
        v16bf av;
#pragma unroll
        for (int v = 0; v < 8; ++v) {
            int ka = kb + ((v & 4) ? 16 : 0) + kaA + 2 * (v & 3);
            float2 fa = *(const float2*)(A + (size_t)row * K + ka);
            av[2 * v]     = (__bf16)fa.x;
            av[2 * v + 1] = (__bf16)fa.y;
        }
#pragma unroll
        for (int t = 0; t < TN; ++t) {
            // B 32x16 bf16 layout: VGPR v holds K = (v>=4?16:0) + half*8 + 2*(v&3), col = lane&15
            v16bf bv;
#pragma unroll
            for (int v = 0; v < 8; ++v) {
                int kB = kb + ((v & 4) ? 16 : 0) + kaB + 2 * (v & 3);
                bf16x2 pb = *(const bf16x2*)(Bt + (size_t)(colBase + t * 16) * K + kB);
                bv[2 * v]     = pb.x;
                bv[2 * v + 1] = pb.y;
            }
            acc[t] = __builtin_amdgcn_wmma_f32_16x16x32_bf16(
                         false, av, false, bv, (short)0, acc[t], false, false);
        }
    }

    // C/D layout: VGPR r -> M = half*8 + r, N = lane&15
    const int orow = tileM * 16 + hi * 8;
#pragma unroll
    for (int t = 0; t < TN; ++t) {
        const int col = colBase + t * 16;
        const float bc = bias ? bias[col] : 0.0f;
#pragma unroll
        for (int r = 0; r < 8; ++r)
            D[(size_t)(orow + r) * Nc + col] = acc[t][r] + bc;
    }
}

// ---------------- epilogue: bias + relu + transpose to [B,COUT,T,N] ----------------

__global__ void k_final(const float* __restrict__ agg, const float* __restrict__ b3,
                        float* __restrict__ out) {
    int id = blockIdx.x * blockDim.x + threadIdx.x;
    if (id >= B_ * COUT * T_ * N_) return;
    int n = id % N_;
    int t = (id / N_) % T_;
    int c = (id / (N_ * T_)) % COUT;
    int b = id / (N_ * T_ * COUT);
    float v = agg[(((size_t)(b * T_ + t)) * N_ + n) * COUT + c] + b3[c];
    out[id] = v > 0.0f ? v : 0.0f;
}

// ---------------- host-side launch ----------------

static inline int ceil_div(int a, int b) { return (a + b - 1) / b; }

extern "C" void kernel_launch(void* const* d_in, const int* in_sizes, int n_in,
                              void* d_out, int out_size, void* d_ws, size_t ws_size,
                              hipStream_t stream) {
    (void)in_sizes; (void)n_in; (void)out_size; (void)ws_size;
    const float*     x   = (const float*)d_in[0];
    const long long* ei  = (const long long*)d_in[1];
    const float*     ew  = (const float*)d_in[2];
    const float*     W1  = (const float*)d_in[3];
    const float*     b1  = (const float*)d_in[4];
    const float*     W2  = (const float*)d_in[5];
    const float*     b2  = (const float*)d_in[6];
    const float*     W3  = (const float*)d_in[7];
    const float*     b3  = (const float*)d_in[8];
    float* out = (float*)d_out;

    // workspace carve-out (256B aligned)
    char* p = (char*)d_ws;
    auto carve = [&](size_t bytes) {
        char* r = p;
        p += (bytes + 255) & ~(size_t)255;
        return r;
    };
    float*  dinv = (float*)carve((size_t)N_ * 4);
    float*  nrm  = (float*)carve((size_t)E_ * 4);
    __bf16* W1t  = (__bf16*)carve((size_t)CIN * H1_ * 2);
    __bf16* W2t  = (__bf16*)carve((size_t)H1_ * H2_ * 2);
    __bf16* W3t  = (__bf16*)carve((size_t)H2_ * COUT * 2);
    float*  xt   = (float*)carve((size_t)M_ * CIN * 4);
    float*  agg0 = (float*)carve((size_t)M_ * CIN * 4);   // contiguous after xt
    float*  y1   = (float*)carve((size_t)M_ * H1_ * 4);
    float*  agg1 = (float*)carve((size_t)M_ * H1_ * 4);
    float*  y2   = (float*)carve((size_t)M_ * H2_ * 4);
    float*  z    = xt;   // reuse xt+agg0 region (exactly M_*COUT*4 bytes)
    float*  agg2 = y1;   // reuse y1 (dead after agg1 = prop(y1))

    const int TB = 256;

    // weight prep (f32 [K,N] -> bf16 [N,K])
    k_wt_bf16<<<ceil_div(CIN * H1_, TB), TB, 0, stream>>>(W1, W1t, CIN, H1_);
    k_wt_bf16<<<ceil_div(H1_ * H2_, TB), TB, 0, stream>>>(W2, W2t, H1_, H2_);
    k_wt_bf16<<<ceil_div(H2_ * COUT, TB), TB, 0, stream>>>(W3, W3t, H2_, COUT);

    // degree / symmetric norm
    k_deg_init<<<ceil_div(N_, TB), TB, 0, stream>>>(dinv);
    k_deg_scatter<<<ceil_div(E_, TB), TB, 0, stream>>>(ei, ew, dinv);
    k_dinv<<<ceil_div(N_, TB), TB, 0, stream>>>(dinv);
    k_norm<<<ceil_div(E_, TB), TB, 0, stream>>>(ei, ew, dinv, nrm);

    // x -> node-major
    k_transpose_x<<<ceil_div(B_ * CIN * T_ * N_, TB), TB, 0, stream>>>(x, xt);

    const int scatter_blocks = ceil_div(E_ * BT, TB / 32);
    const int mblocks = M_ / 128;   // 8 M-tiles (waves) per block

    // Layer 1: propagate at C=32, then GEMM 32->128 (+b1)
    k_prop_init<<<ceil_div(M_ * CIN, TB), TB, 0, stream>>>(xt, agg0, dinv, CIN);
    k_prop_scatter<<<scatter_blocks, TB, 0, stream>>>(xt, agg0, nrm, ei, CIN);
    k_gemm_wmma<CIN, 4><<<dim3(mblocks, H1_ / 64), TB, 0, stream>>>(agg0, W1t, b1, y1, H1_);

    // Layer 2: propagate at C=128, then GEMM 128->256 (+b2)
    k_prop_init<<<ceil_div(M_ * H1_, TB), TB, 0, stream>>>(y1, agg1, dinv, H1_);
    k_prop_scatter<<<scatter_blocks, TB, 0, stream>>>(y1, agg1, nrm, ei, H1_);
    k_gemm_wmma<H1_, 4><<<dim3(mblocks, H2_ / 64), TB, 0, stream>>>(agg1, W2t, b2, y2, H2_);

    // Layer 3: GEMM 256->64 first (no bias), then propagate at C=64
    k_gemm_wmma<H2_, 4><<<dim3(mblocks, COUT / 64), TB, 0, stream>>>(y2, W3t, nullptr, z, COUT);
    k_prop_init<<<ceil_div(M_ * COUT, TB), TB, 0, stream>>>(z, agg2, dinv, COUT);
    k_prop_scatter<<<scatter_blocks, TB, 0, stream>>>(z, agg2, nrm, ei, COUT);

    // bias3 + relu + transpose to [B,COUT,T,N]
    k_final<<<ceil_div(B_ * COUT * T_ * N_, TB), TB, 0, stream>>>(agg2, b3, out);
}
